// FusedThorExpert_52304111730966
// MI455X (gfx1250) — matrix operand order repaired and checked by reading, not compiled, not run on hardware
//
#include <hip/hip_runtime.h>

// MI455X / gfx1250, wave32.
// Pipeline: [convert f32->bf16 pre-pass] -> [bf16 WMMA GEMM1 + exact GELU -> bf16 h]
//           -> [bf16 WMMA GEMM2 -> f32 out]
// GEMMs: 128x128 block tile, 4 waves (wave tile 64x64 = 4x4 WMMA 16x16x32),
// triple-buffered LDS staged via CDNA5 async global->LDS copies (ASYNCcnt),
// one workgroup barrier per K-step, branch-free steady-state loop.

typedef __attribute__((ext_vector_type(16))) __bf16 v16bf;
typedef __attribute__((ext_vector_type(8)))  float  v8f;

__device__ __forceinline__ unsigned int pack2_bf16(float lo, float hi) {
  unsigned int ul = __builtin_bit_cast(unsigned int, lo);
  unsigned int uh = __builtin_bit_cast(unsigned int, hi);
  ul = (ul + 0x7FFFu + ((ul >> 16) & 1u)) >> 16;   // round-to-nearest-even
  uh = (uh + 0x7FFFu + ((uh >> 16) & 1u)) >> 16;
  return (uh << 16) | (ul & 0xFFFFu);
}

__device__ __forceinline__ unsigned short to_bf16(float f) {
  unsigned int u = __builtin_bit_cast(unsigned int, f);
  return (unsigned short)((u + 0x7FFFu + ((u >> 16) & 1u)) >> 16);
}

// ---------------- pre-pass: fp32 -> bf16, 8 elements/thread ----------------
__global__ __launch_bounds__(256)
void convert_bf16(const float* __restrict__ in, unsigned short* __restrict__ out,
                  long long n) {
  long long i = ((long long)blockIdx.x * blockDim.x + threadIdx.x) * 8;
  if (i + 8 <= n) {
    const float4* p = (const float4*)(in + i);
    float4 f0 = p[0], f1 = p[1];
    uint4 q;
    q.x = pack2_bf16(f0.x, f0.y); q.y = pack2_bf16(f0.z, f0.w);
    q.z = pack2_bf16(f1.x, f1.y); q.w = pack2_bf16(f1.z, f1.w);
    *(uint4*)(out + i) = q;
  }
}

// ---- async global->LDS staging: one 64B row per matrix per thread = 4 x b128 ----
// (the instruction offset applies to BOTH the global and LDS addresses)
__device__ __forceinline__ void stage_async(const void* ga, const void* gb,
                                            unsigned la, unsigned lb) {
  asm volatile("global_load_async_to_lds_b128 %0, %1, off"
               :: "v"(la), "v"(ga) : "memory");
  asm volatile("global_load_async_to_lds_b128 %0, %1, off offset:16"
               :: "v"(la), "v"(ga) : "memory");
  asm volatile("global_load_async_to_lds_b128 %0, %1, off offset:32"
               :: "v"(la), "v"(ga) : "memory");
  asm volatile("global_load_async_to_lds_b128 %0, %1, off offset:48"
               :: "v"(la), "v"(ga) : "memory");
  asm volatile("global_load_async_to_lds_b128 %0, %1, off"
               :: "v"(lb), "v"(gb) : "memory");
  asm volatile("global_load_async_to_lds_b128 %0, %1, off offset:16"
               :: "v"(lb), "v"(gb) : "memory");
  asm volatile("global_load_async_to_lds_b128 %0, %1, off offset:32"
               :: "v"(lb), "v"(gb) : "memory");
  asm volatile("global_load_async_to_lds_b128 %0, %1, off offset:48"
               :: "v"(lb), "v"(gb) : "memory");
}

#define LDS_ROW_Q 5          // 32 bf16 (64B) + 16B pad = 80B = 5 uint4 per row
#define BUF_Q (128 * LDS_ROW_Q)
#define BUF_BYTES (BUF_Q * 16)

// C[e] = act(A[e] @ B[e]^T + bias[e]);  A: MxK bf16, B: NxK bf16 (K-contiguous)
template <bool DO_GELU, bool OUT_BF16>
__global__ __launch_bounds__(128)
void expert_gemm_bf16(const unsigned short* __restrict__ Aall,
                      const unsigned short* __restrict__ Ball,
                      const float* __restrict__ biasAll, void* __restrict__ OutAll,
                      int M, int N, int K, int mBlocks, int nBlocks) {
  __shared__ uint4 As[3][BUF_Q];   // 3 x 10240B
  __shared__ uint4 Bs[3][BUF_Q];   // 3 x 10240B  (total 60KB)

  const int bpe = mBlocks * nBlocks;
  const int e   = blockIdx.x / bpe;
  const int rem = blockIdx.x % bpe;
  const int mBase = (rem / nBlocks) * 128;
  const int nBase = (rem % nBlocks) * 128;

  const int tid  = threadIdx.x;      // 0..127
  const int lane = tid & 31;
  const int half = lane >> 4;
  const int lrow = lane & 15;
  const int w    = tid >> 5;         // wave 0..3
  const int wr   = w >> 1;           // 0..1 : 64-row slab
  const int wc   = w & 1;            // 0..1 : 64-col slab

  // staging: each thread owns one full 32-element K-row of A and of B
  const unsigned short* aG = Aall + (size_t)e * M * K + (size_t)(mBase + tid) * K;
  const unsigned short* bG = Ball + (size_t)e * N * K + (size_t)(nBase + tid) * K;
  const float* bias = biasAll + (size_t)e * N;

  // LDS byte addresses (generic-pointer low 32 bits == LDS offset)
  const unsigned ldsA = (unsigned)(unsigned long long)&As[0][tid * LDS_ROW_Q];
  const unsigned ldsB = (unsigned)(unsigned long long)&Bs[0][tid * LDS_ROW_Q];

  v8f acc[4][4];
#pragma unroll
  for (int mi = 0; mi < 4; ++mi)
#pragma unroll
    for (int ni = 0; ni < 4; ++ni)
#pragma unroll
      for (int r = 0; r < 8; ++r) acc[mi][ni][r] = 0.0f;

  union FragU { uint4 q[2]; v16bf v; };

  const int steps = K >> 5;          // >= 2 always (16 or 32)
  stage_async(aG, bG, ldsA, ldsB);   // prologue: k-slab 0 -> buf 0

  int rb = 0;
  for (int s = 0; s < steps - 1; ++s) {
    int wb = rb + 1; if (wb == 3) wb = 0;
    const int k = (s + 1) << 5;
    stage_async(aG + k, bG + k, ldsA + (unsigned)wb * BUF_BYTES,
                                ldsB + (unsigned)wb * BUF_BYTES);
    // the 8 just-issued ops may remain outstanding; stage s retired (in-order)
    asm volatile("s_wait_asynccnt 0x8" ::: "memory");
    __syncthreads();

    const uint4* Asb = As[rb];
    const uint4* Bsb = Bs[rb];
    FragU bf[4];
#pragma unroll
    for (int ni = 0; ni < 4; ++ni) {
      const int n = wc * 64 + ni * 16 + lrow;
      bf[ni].q[0] = Bsb[n * LDS_ROW_Q + half];
      bf[ni].q[1] = Bsb[n * LDS_ROW_Q + half + 2];
    }
#pragma unroll
    for (int mi = 0; mi < 4; ++mi) {
      FragU af;
      const int r = wr * 64 + mi * 16 + lrow;
      af.q[0] = Asb[r * LDS_ROW_Q + half];
      af.q[1] = Asb[r * LDS_ROW_Q + half + 2];
#pragma unroll
      for (int ni = 0; ni < 4; ++ni)
        acc[mi][ni] = __builtin_amdgcn_wmma_f32_16x16x32_bf16(
            false, af.v, false, bf[ni].v, (short)0, acc[mi][ni], false, false);
    }
    rb = wb;
  }

  // ---- peeled final step ----
  asm volatile("s_wait_asynccnt 0x0" ::: "memory");
  __syncthreads();
  {
    const uint4* Asb = As[rb];
    const uint4* Bsb = Bs[rb];
    FragU bf[4];
#pragma unroll
    for (int ni = 0; ni < 4; ++ni) {
      const int n = wc * 64 + ni * 16 + lrow;
      bf[ni].q[0] = Bsb[n * LDS_ROW_Q + half];
      bf[ni].q[1] = Bsb[n * LDS_ROW_Q + half + 2];
    }
#pragma unroll
    for (int mi = 0; mi < 4; ++mi) {
      FragU af;
      const int r = wr * 64 + mi * 16 + lrow;
      af.q[0] = Asb[r * LDS_ROW_Q + half];
      af.q[1] = Asb[r * LDS_ROW_Q + half + 2];
#pragma unroll
      for (int ni = 0; ni < 4; ++ni)
        acc[mi][ni] = __builtin_amdgcn_wmma_f32_16x16x32_bf16(
            false, af.v, false, bf[ni].v, (short)0, acc[mi][ni], false, false);
    }
  }

  // ---- epilogue: bias, exact GELU, store per C/D layout ----
  const size_t outStride = (size_t)M * N;
#pragma unroll
  for (int ni = 0; ni < 4; ++ni) {
    const int gcol = nBase + wc * 64 + ni * 16 + lrow;
    const float bv = bias[gcol];
#pragma unroll
    for (int mi = 0; mi < 4; ++mi) {
#pragma unroll
      for (int r = 0; r < 8; ++r) {
        float v = acc[mi][ni][r] + bv;
        if constexpr (DO_GELU)
          v = 0.5f * v * (1.0f + erff(v * 0.70710678118654752f));
        const int grow = mBase + wr * 64 + mi * 16 + half * 8 + r;
        if constexpr (OUT_BF16) {
          unsigned short* o = (unsigned short*)OutAll + (size_t)e * outStride;
          o[(size_t)grow * N + gcol] = to_bf16(v);
        } else {
          float* o = (float*)OutAll + (size_t)e * outStride;
          o[(size_t)grow * N + gcol] = v;
        }
      }
    }
  }
}

extern "C" void kernel_launch(void* const* d_in, const int* in_sizes, int n_in,
                              void* d_out, int out_size, void* d_ws, size_t ws_size,
                              hipStream_t stream) {
  // inputs: 0:inter_state f32 [16384,512], 1:capacities i64 (fixed 1024),
  //         2:W1 f32 [16,1024,512], 3:b1 [16,1024], 4:W2 f32 [16,512,1024], 5:b2 [16,512]
  const float* x  = (const float*)d_in[0];
  const float* W1 = (const float*)d_in[2];
  const float* b1 = (const float*)d_in[3];
  const float* W2 = (const float*)d_in[4];
  const float* b2 = (const float*)d_in[5];

  const long long nX = 16LL * 1024 * 1024 / 2 * 1;            // 16384*512 = 8,388,608
  const long long nW = 16LL * 1024 * 512;                     // 8,388,608 per weight
  unsigned short* h   = (unsigned short*)d_ws;                // 16,777,216 bf16 (32 MB)
  unsigned short* Xb  = h   + 16LL * 1024 * 1024;             // +16 MB
  unsigned short* W1b = Xb  + nX;                             // +16 MB
  unsigned short* W2b = W1b + nW;                             // +16 MB (80 MB ws total)

  convert_bf16<<<dim3(4096), dim3(256), 0, stream>>>(x,  Xb,  nX);
  convert_bf16<<<dim3(4096), dim3(256), 0, stream>>>(W1, W1b, nW);
  convert_bf16<<<dim3(4096), dim3(256), 0, stream>>>(W2, W2b, nW);

  // GEMM1 + GELU: per expert M=1024, N=I=1024, K=H=512 -> 8x8 blocks/expert
  expert_gemm_bf16<true, true><<<dim3(16 * 8 * 8), dim3(128), 0, stream>>>(
      Xb, W1b, b1, (void*)h, 1024, 1024, 512, 8, 8);
  // GEMM2: per expert M=1024, N=H=512, K=I=1024 -> 8x4 blocks/expert
  expert_gemm_bf16<false, false><<<dim3(16 * 8 * 4), dim3(128), 0, stream>>>(
      h, W2b, b2, d_out, 1024, 512, 1024, 8, 4);
}
